// DeformableAttention_43482248905333
// MI455X (gfx1250) — matrix operand order repaired and checked by reading, not compiled, not run on hardware
//
#include <hip/hip_runtime.h>
#include <hip/hip_bf16.h>
#include <stdint.h>

typedef __attribute__((ext_vector_type(16))) _Float16 v16h;
typedef __attribute__((ext_vector_type(8)))  float    v8f;

#define B_      8
#define NQ_     5440
#define C_      256
#define HEADS_  8
#define LEVELS_ 4
#define POINTS_ 4
#define HD_     32
#define NPROJ_  384           // 256 (offsets) + 128 (attn logits)
#define M_      (B_ * NQ_)    // 43520 rows
#define KDIM_   256           // K of every GEMM here

// ---------------------------------------------------------------- conversions
__global__ void cvt_f32_to_f16(const float* __restrict__ src,
                               _Float16* __restrict__ dst, int n) {
  int i = blockIdx.x * blockDim.x + threadIdx.x;
  if (i < n) dst[i] = (_Float16)src[i];
}

// WTproj[384][256] = [w_off^T ; w_attn^T] in f16; WTout[256][256] = w_out^T in f16
__global__ void build_wt(const float* __restrict__ w_off,
                         const float* __restrict__ w_attn,
                         const float* __restrict__ w_out,
                         _Float16* __restrict__ wtproj,
                         _Float16* __restrict__ wtout) {
  int i = blockIdx.x * blockDim.x + threadIdx.x;
  if (i < NPROJ_ * C_) {
    int n = i / C_, k = i % C_;
    float v = (n < 256) ? w_off[k * 256 + n] : w_attn[k * 128 + (n - 256)];
    wtproj[i] = (_Float16)v;
  } else {
    int j = i - NPROJ_ * C_;
    if (j < C_ * C_) {
      int n = j / C_, k = j % C_;
      wtout[j] = (_Float16)w_out[k * C_ + n];
    }
  }
}

// ---------------------------------------------------------------- WMMA GEMM
// Cout[M][N] = A[M][K=256] (f16 row-major) * BT[N][K]^T (f16) + bias[N]
// Block = (32,4) = 4 waves. Block tile: 128 rows x 64 cols.
//   - B tile (64 x 256 f16 = 32 KB) staged in LDS once per block, shared by all
//     4 waves; steady-state loop reads B via ds_load_b128.
//   - Each wave computes 32 rows x 64 cols: 2 A fragments x 4 B fragments
//     -> 8 v_wmma per k-step, B fragment reused across the 2 M-subtiles.
// grid = (M/128, N/64). M=43520, K=256, N in {384,256}: all exact.
__global__ void wmma_gemm(const _Float16* __restrict__ A,
                          const _Float16* __restrict__ BT,
                          const float* __restrict__ bias,
                          float* __restrict__ Cout,
                          int M, int N) {
  constexpr int K = KDIM_;
  __shared__ __align__(16) _Float16 Bs[64 * K];   // 32 KB

  const int lane = threadIdx.x & 31;
  const int half = lane >> 4;     // 0: lanes 0-15, 1: lanes 16-31
  const int ml   = lane & 15;     // A row / B col within a 16x16 tile
  const int wv   = threadIdx.y;   // wave id 0..3
  const int n0   = blockIdx.y * 64;
  const int m0   = blockIdx.x * 128 + wv * 32;

  // ---- cooperative stage of B tile: BT[n0 .. n0+63][0..255] -> LDS (flat copy)
  {
    const uint4* src = (const uint4*)(BT + (size_t)n0 * K);
    uint4* dst = (uint4*)Bs;
    const int tid = wv * 32 + lane;                 // 0..127
    constexpr int CHUNKS = 64 * K * (int)sizeof(_Float16) / 16;  // 2048
#pragma unroll
    for (int i = 0; i < CHUNKS / 128; ++i)
      dst[tid + i * 128] = src[tid + i * 128];
  }
  __syncthreads();

  v8f acc0[4] = {};
  v8f acc1[4] = {};

  const _Float16* Arow0 = A + (size_t)(m0 + ml) * K;
  const _Float16* Arow1 = A + (size_t)(m0 + 16 + ml) * K;

  for (int kb = 0; kb < K; kb += 32) {
    // A fragment: per lane two contiguous 16B runs at halves
    //   [kb + 8*half, +8) and [kb + 16 + 8*half, +8)
    union { v16h h; uint4 q[2]; } a0, a1;
    {
      const uint4* p0 = (const uint4*)(Arow0 + kb + (half << 3));
      const uint4* p1 = (const uint4*)(Arow1 + kb + (half << 3));
      a0.q[0] = p0[0]; a0.q[1] = p0[2];   // +2 uint4 = +16 halves
      a1.q[0] = p1[0]; a1.q[1] = p1[2];
    }
#pragma unroll
    for (int t = 0; t < 4; ++t) {
      // B fragment: two 16B runs at halves [kb + 8*half, +8) and [kb + 16 + 8*half, +8)
      union { v16h h; uint4 q[2]; } b;
      const uint4* pb = (const uint4*)(Bs + (size_t)(t * 16 + ml) * K + kb + (half << 3));
      b.q[0] = pb[0]; b.q[1] = pb[2];
      acc0[t] = __builtin_amdgcn_wmma_f32_16x16x32_f16(
          false, a0.h, false, b.h, (short)0, acc0[t], false, false);
      acc1[t] = __builtin_amdgcn_wmma_f32_16x16x32_f16(
          false, a1.h, false, b.h, (short)0, acc1[t], false, false);
    }
  }

#pragma unroll
  for (int t = 0; t < 4; ++t) {
    const int col = n0 + t * 16 + ml;
    const float bv = bias ? bias[col] : 0.0f;
#pragma unroll
    for (int r = 0; r < 8; ++r) {
      const int row = r + (half << 3);      // C/D layout: M = r + 8*half, N = lane&15
      Cout[(size_t)(m0 + row) * N + col]      = acc0[t][r] + bv;
      Cout[(size_t)(m0 + 16 + row) * N + col] = acc1[t][r] + bv;
    }
  }
}

// ---------------------------------------------------------------- sampling
// One block per (b,q); thread = h*32 + d. Softmax over 16 attn logits per head,
// bilinear gather over 4 levels x 4 points, write f16 sampled (b,q,h*hd+d).
__global__ void deform_sample(const float* __restrict__ refp,
                              const float* __restrict__ feat,
                              const float* __restrict__ S,
                              const float* __restrict__ b_off,
                              const float* __restrict__ b_attn,
                              _Float16* __restrict__ outh) {
  const int bq  = blockIdx.x;
  const int b   = bq / NQ_;
  const int tid = threadIdx.x;
  const int h   = tid >> 5;
  const int d   = tid & 31;
  const float* Srow = S + (size_t)bq * NPROJ_;

  // softmax over this head's 16 (level,point) logits
  float w[16];
  float mx = -1e30f;
#pragma unroll
  for (int j = 0; j < 16; ++j) {
    float v = Srow[256 + h * 16 + j] + b_attn[h * 16 + j];
    w[j] = v;
    mx = fmaxf(mx, v);
  }
  float sum = 0.0f;
#pragma unroll
  for (int j = 0; j < 16; ++j) { w[j] = __expf(w[j] - mx); sum += w[j]; }
  const float inv = 1.0f / sum;

  const float rx = refp[bq * 2 + 0];
  const float ry = refp[bq * 2 + 1];

  const int Wl_[4] = {64, 32, 16, 8};
  const int Hl_[4] = {64, 32, 16, 8};
  const int st_[4] = {0, 4096, 5120, 5376};

  float acc = 0.0f;
#pragma unroll
  for (int l = 0; l < 4; ++l) {
    const int Wl = Wl_[l], Hl = Hl_[l];
    const float* fbase = feat + ((size_t)b * NQ_ + st_[l]) * C_ + h * HD_ + d;
#pragma unroll
    for (int p = 0; p < 4; ++p) {
      const int j = l * 4 + p;
      const int oidx = ((h * LEVELS_ + l) * POINTS_ + p) * 2;
      const float ox = Srow[oidx]     + b_off[oidx];
      const float oy = Srow[oidx + 1] + b_off[oidx + 1];
      // x = loc_x*W - 0.5 = rx*W + ox - 0.5 ; y = ry*H + oy - 0.5
      const float x = rx * (float)Wl + ox - 0.5f;
      const float y = ry * (float)Hl + oy - 0.5f;
      const float xf = floorf(x), yf = floorf(y);
      const int   x0 = (int)xf,   y0 = (int)yf;
      const float fx = x - xf,    fy = y - yf;
      const float aw = w[j] * inv;
#pragma unroll
      for (int dy = 0; dy < 2; ++dy) {
        const int yi = y0 + dy;
        if (yi < 0 || yi > Hl - 1) continue;
        const float wy = dy ? fy : (1.0f - fy);
#pragma unroll
        for (int dx = 0; dx < 2; ++dx) {
          const int xi = x0 + dx;
          if (xi < 0 || xi > Wl - 1) continue;
          const float wx = dx ? fx : (1.0f - fx);
          acc += aw * wy * wx * fbase[(size_t)(yi * Wl + xi) * C_];
        }
      }
    }
  }
  outh[(size_t)bq * C_ + tid] = (_Float16)acc;
}

// ---------------------------------------------------------------- launch
extern "C" void kernel_launch(void* const* d_in, const int* in_sizes, int n_in,
                              void* d_out, int out_size, void* d_ws, size_t ws_size,
                              hipStream_t stream) {
  (void)in_sizes; (void)n_in; (void)out_size; (void)ws_size;
  const float* query  = (const float*)d_in[0];
  const float* refp   = (const float*)d_in[1];
  const float* feat   = (const float*)d_in[2];
  // d_in[3] spatial_shapes / d_in[4] level_start_index are compile-time constants
  const float* w_off  = (const float*)d_in[5];
  const float* b_off  = (const float*)d_in[6];
  const float* w_attn = (const float*)d_in[7];
  const float* b_attn = (const float*)d_in[8];
  const float* w_out  = (const float*)d_in[9];
  const float* b_out  = (const float*)d_in[10];
  float* out = (float*)d_out;

  char* ws = (char*)d_ws;
  size_t cur = 0;
  auto alloc = [&](size_t bytes) -> void* {
    cur = (cur + 255) & ~(size_t)255;
    void* p = ws + cur;
    cur += bytes;
    return p;
  };
  _Float16* Aq  = (_Float16*)alloc((size_t)M_ * C_ * sizeof(_Float16));      // f16 query
  _Float16* WTp = (_Float16*)alloc((size_t)NPROJ_ * C_ * sizeof(_Float16));  // [w_off|w_attn]^T
  _Float16* WTo = (_Float16*)alloc((size_t)C_ * C_ * sizeof(_Float16));      // w_out^T
  float*    S   = (float*)   alloc((size_t)M_ * NPROJ_ * sizeof(float));     // proj output f32
  _Float16* Sm  = (_Float16*)alloc((size_t)M_ * C_ * sizeof(_Float16));      // sampled f16

  // 1) query -> f16
  {
    int n = M_ * C_;
    cvt_f32_to_f16<<<(n + 255) / 256, 256, 0, stream>>>(query, Aq, n);
  }
  // 2) transpose + convert weights
  {
    int n = NPROJ_ * C_ + C_ * C_;
    build_wt<<<(n + 255) / 256, 256, 0, stream>>>(w_off, w_attn, w_out, WTp, WTo);
  }
  // 3) fused projection GEMM: S = query @ [w_off | w_attn]  (biases folded later)
  {
    dim3 grid(M_ / 128, NPROJ_ / 64);
    dim3 block(32, 4);
    wmma_gemm<<<grid, block, 0, stream>>>(Aq, WTp, nullptr, S, M_, NPROJ_);
  }
  // 4) softmax + bilinear gather -> f16 sampled features
  {
    deform_sample<<<M_, 256, 0, stream>>>(refp, feat, S, b_off, b_attn, Sm);
  }
  // 5) output GEMM: out = sampled @ w_out + b_out (f32 result)
  {
    dim3 grid(M_ / 128, C_ / 64);
    dim3 block(32, 4);
    wmma_gemm<<<grid, block, 0, stream>>>(Sm, WTo, b_out, out, M_, C_);
  }
}
